// SVGTransformerLM_MuP_69621419868559
// MI455X (gfx1250) — matrix operand order repaired and checked by reading, not compiled, not run on hardware
//
#include <hip/hip_runtime.h>
#include <hip/hip_bf16.h>
#include <math.h>

// ---------------------------------------------------------------------------
// Model dims (compile-time constants matching the reference)
// ---------------------------------------------------------------------------
#define VOCAB 32000
#define DMODEL 1024
#define NLAYER 6
#define NHEAD 16
#define FFN 4096
#define SEQ 1024
#define BATCH 2
#define TLEN 1024
#define DHEAD 64
#define MROWS (BATCH * TLEN)   // 2048

typedef __attribute__((ext_vector_type(16))) __bf16 v16bf;
typedef __attribute__((ext_vector_type(8)))  float  v8f;

union Frag16 {       // 16 bf16 halves == 32 bytes == two uint4
    v16bf v;
    uint4 q[2];
};

static __device__ __forceinline__ v8f wmma_bf16(const v16bf a, const v16bf b, const v8f c) {
    // D = A(16x32 bf16) * B(32x16 bf16) + C(16x16 f32)
    return __builtin_amdgcn_wmma_f32_16x16x32_bf16(false, a, false, b, (short)0, c, false, false);
}

// Flat pointer to LDS byte offset (ISA 10.2: LDS flat addr carries offset in [31:0])
static __device__ __forceinline__ unsigned lds_off(const void* p) {
    return (unsigned)(uintptr_t)p;
}

// ---------------------------------------------------------------------------
// Embedding: x[b,t,:] = tok_emb[idx[b,t]] + pos_emb[t]
// ---------------------------------------------------------------------------
__global__ __launch_bounds__(256) void embed_kernel(
    const int* __restrict__ idx, const float* __restrict__ tok,
    const float* __restrict__ pos, float* __restrict__ x)
{
    size_t i = (size_t)blockIdx.x * 256 + threadIdx.x;   // over B*T*D
    int d = (int)(i % DMODEL);
    size_t bt = i / DMODEL;
    int t = (int)(bt % TLEN);
    x[i] = tok[(size_t)idx[bt] * DMODEL + d] + pos[(size_t)t * DMODEL + d];
}

// ---------------------------------------------------------------------------
// LayerNorm (biased variance, eps 1e-5), emits bf16 activations
// one row (D=1024) per 256-thread block
// ---------------------------------------------------------------------------
__global__ __launch_bounds__(256) void layernorm_kernel(
    const float* __restrict__ x, const float* __restrict__ w,
    const float* __restrict__ b, __bf16* __restrict__ out, int Dd)
{
    __shared__ float redS[8];
    __shared__ float redQ[8];
    const int row = blockIdx.x;
    const float* xr = x + (size_t)row * Dd;
    float s = 0.f, ss = 0.f;
    for (int i = threadIdx.x; i < Dd; i += 256) {
        float v = xr[i];
        s += v; ss += v * v;
    }
    for (int m = 16; m >= 1; m >>= 1) {
        s  += __shfl_xor(s,  m, 32);
        ss += __shfl_xor(ss, m, 32);
    }
    const int lane = threadIdx.x & 31, wid = threadIdx.x >> 5;
    if (lane == 0) { redS[wid] = s; redQ[wid] = ss; }
    __syncthreads();
    float ts = 0.f, tq = 0.f;
    #pragma unroll
    for (int i = 0; i < 8; i++) { ts += redS[i]; tq += redQ[i]; }
    const float mean = ts / (float)Dd;
    const float var  = tq / (float)Dd - mean * mean;
    const float rinv = rsqrtf(var + 1e-5f);
    for (int i = threadIdx.x; i < Dd; i += 256) {
        float v = (xr[i] - mean) * rinv * w[i] + b[i];
        out[(size_t)row * Dd + i] = (__bf16)v;
    }
}

// ---------------------------------------------------------------------------
// GEMM: out[M,N] = epilogue( A[M,K](bf16) @ W[N,K]^T(fp32->bf16) )
// 256 threads / 8 waves, workgroup tile 128x256, wave tile 64x64, K-step 32.
// A tile staged via GLOBAL_LOAD_ASYNC_TO_LDS_B128 (ASYNCcnt path);
// W tile staged via load + v_cvt_pk_bf16_f32 + ds_store.
// flags: 1=+bias[N], 2=gelu(exact erf), 4=+residual[M,N]
// ---------------------------------------------------------------------------
#define BM 128
#define BN 256
#define BK 32

__global__ __launch_bounds__(256) void gemm_bf16_kernel(
    const __bf16* __restrict__ A, const float* __restrict__ W,
    const float* __restrict__ bias, const float* __restrict__ resid,
    float* __restrict__ outF, __bf16* __restrict__ outB,
    int Md, int Nd, int Kd, int flags)
{
    __shared__ __align__(16) __bf16 As[BM][BK];   //  8 KiB
    __shared__ __align__(16) __bf16 Ws[BN][BK];   // 16 KiB

    const int tid   = threadIdx.x;
    const int lane  = tid & 31;
    const int wid   = tid >> 5;       // 0..7
    const int waveM = wid >> 2;       // 0..1  -> 64-row slab
    const int waveN = wid & 3;        // 0..3  -> 64-col slab
    const int bm    = blockIdx.x * BM;
    const int bn    = blockIdx.y * BN;
    const int lrow  = lane & 15;
    const int lhi   = lane >> 4;

    v8f acc[4][4];
    #pragma unroll
    for (int i = 0; i < 4; i++)
        #pragma unroll
        for (int j = 0; j < 4; j++) acc[i][j] = (v8f){0.f,0.f,0.f,0.f,0.f,0.f,0.f,0.f};

    // A staging assignment: 2 threads per row, 16 halves (32 B) each
    const int ar = tid >> 1;
    const int ac = (tid & 1) * 16;
    const unsigned ldsA = lds_off(&As[ar][ac]);
    // W staging assignment: 1 thread per row, 32 floats -> 32 bf16
    const int wr = tid;

    for (int k0 = 0; k0 < Kd; k0 += BK) {
        __syncthreads();
        // --- A tile: async global->LDS copy (2 x 16 B per thread) ---
        {
            const unsigned long long ga =
                (unsigned long long)(uintptr_t)(A + (size_t)(bm + ar) * Kd + k0 + ac);
            asm volatile(
                "global_load_async_to_lds_b128 %0, %1, off\n\t"
                "global_load_async_to_lds_b128 %0, %1, off offset:16"
                :: "v"(ldsA), "v"(ga) : "memory");
        }
        // --- W tile: fp32 -> bf16 through VGPRs ---
        {
            const float* src = W + (size_t)(bn + wr) * Kd + k0;
            union { __bf16 h[32]; uint4 q[4]; } wbuf;
            #pragma unroll
            for (int j = 0; j < 32; j += 4) {
                float4 f = *(const float4*)(src + j);
                wbuf.h[j + 0] = (__bf16)f.x;
                wbuf.h[j + 1] = (__bf16)f.y;
                wbuf.h[j + 2] = (__bf16)f.z;
                wbuf.h[j + 3] = (__bf16)f.w;
            }
            #pragma unroll
            for (int u = 0; u < 4; u++) *(uint4*)(&Ws[wr][u * 8]) = wbuf.q[u];
            if (k0 + BK < Kd) __builtin_prefetch(src + BK, 0, 1);
        }
        // async copies must land before the barrier releases readers
        asm volatile("s_wait_asynccnt 0x0" ::: "memory");
        __syncthreads();

        // --- gather fragments per ISA VGPR layout, 16 WMMA per K-step ---
        Frag16 a[4];
        #pragma unroll
        for (int i = 0; i < 4; i++) {
            const int m = waveM * 64 + i * 16 + lrow;
            a[i].q[0] = *(const uint4*)(&As[m][lhi * 8]);       // K = khalf*8 + 0..7
            a[i].q[1] = *(const uint4*)(&As[m][16 + lhi * 8]);  // K = 16 + khalf*8 + 0..7
        }
        #pragma unroll
        for (int j = 0; j < 4; j++) {
            Frag16 b;
            const int n = waveN * 64 + j * 16 + lrow;
            b.q[0] = *(const uint4*)(&Ws[n][lhi * 8]);          // K = khalf*8 + 0..7
            b.q[1] = *(const uint4*)(&Ws[n][16 + lhi * 8]);     // K = 16 + khalf*8 + 0..7
            #pragma unroll
            for (int i = 0; i < 4; i++)
                acc[i][j] = wmma_bf16(a[i].v, b.v, acc[i][j]);
        }
    }

    // --- epilogue: C layout VGPR e -> m = e + (lane>>4)*8, n = lane&15 ---
    #pragma unroll
    for (int i = 0; i < 4; i++) {
        #pragma unroll
        for (int j = 0; j < 4; j++) {
            #pragma unroll
            for (int e = 0; e < 8; e++) {
                const int row = bm + waveM * 64 + i * 16 + e + lhi * 8;
                const int col = bn + waveN * 64 + j * 16 + lrow;
                float v = acc[i][j][e];
                if (flags & 1) v += bias[col];
                if (flags & 2) v = 0.5f * v * (1.0f + erff(v * 0.70710678118654752f));
                if (flags & 4) v += resid[(size_t)row * Nd + col];
                if (outF) outF[(size_t)row * Nd + col] = v;
                if (outB) outB[(size_t)row * Nd + col] = (__bf16)v;
            }
        }
    }
}

// ---------------------------------------------------------------------------
// Causal attention, muP 1/dh scaling. One wave per (b, h, 16-query tile).
// Scores via WMMA (Q 16x64 @ K^T), softmax in LDS, P@V via WMMA.
// ---------------------------------------------------------------------------
__global__ __launch_bounds__(32) void attention_kernel(
    const __bf16* __restrict__ q, const __bf16* __restrict__ k,
    const __bf16* __restrict__ v, __bf16* __restrict__ y)
{
    __shared__ __align__(16) float  S [16][TLEN];     // 64 KiB
    __shared__ __align__(16) __bf16 Pb[16][TLEN];     // 32 KiB
    __shared__ __align__(16) __bf16 Vt[DHEAD][32];    //  4 KiB (transposed V tile)

    const int qt    = blockIdx.x;        // 0..T/16-1
    const int bh    = blockIdx.y;        // 0..B*H-1
    const int batch = bh / NHEAD;
    const int head  = bh % NHEAD;
    const int qt0   = qt * 16;
    const int lane  = threadIdx.x;
    const int lrow  = lane & 15;
    const int lhi   = lane >> 4;

    const __bf16* qbase = q + (size_t)batch * TLEN * DMODEL + head * DHEAD;
    const __bf16* kbase = k + (size_t)batch * TLEN * DMODEL + head * DHEAD;
    const __bf16* vbase = v + (size_t)batch * TLEN * DMODEL + head * DHEAD;

    // ---- Q fragments (A operand, 16x32 each; two K-chunks over dh=64) ----
    Frag16 aq[2];
    {
        const __bf16* qr = qbase + (size_t)(qt0 + lrow) * DMODEL;
        aq[0].q[0] = *(const uint4*)(qr + lhi * 8);
        aq[0].q[1] = *(const uint4*)(qr + 16 + lhi * 8);
        aq[1].q[0] = *(const uint4*)(qr + 32 + lhi * 8);
        aq[1].q[1] = *(const uint4*)(qr + 48 + lhi * 8);
    }

    const int nkt = qt + 1;              // causal: key tiles 0..qt

    // ---- scores: S[16][16*nkt] ----
    for (int kt = 0; kt < nkt; kt++) {
        const int kt0 = kt * 16;
        Frag16 bk[2];                    // B operand 32x16 (dh-chunk x keys)
        {
            const __bf16* kr = kbase + (size_t)(kt0 + lrow) * DMODEL;
            bk[0].q[0] = *(const uint4*)(kr + lhi * 8);
            bk[0].q[1] = *(const uint4*)(kr + 16 + lhi * 8);
            bk[1].q[0] = *(const uint4*)(kr + 32 + lhi * 8);
            bk[1].q[1] = *(const uint4*)(kr + 48 + lhi * 8);
        }
        v8f c = (v8f){0.f,0.f,0.f,0.f,0.f,0.f,0.f,0.f};
        c = wmma_bf16(aq[0].v, bk[0].v, c);
        c = wmma_bf16(aq[1].v, bk[1].v, c);
        #pragma unroll
        for (int e = 0; e < 8; e++) {
            const int m = e + lhi * 8;
            const int n = lrow;
            float sc = c[e] * (1.0f / (float)DHEAD);       // muP 1/dh
            if (kt0 + n > qt0 + m) sc = -__builtin_inff(); // causal mask
            S[m][kt0 + n] = sc;
        }
    }
    __syncthreads();

    // ---- softmax per query row ----
    const int ncol = qt0 + 16;
    for (int rr = 0; rr < 16; rr++) {
        float mx = -__builtin_inff();
        for (int cidx = lane; cidx < ncol; cidx += 32) mx = fmaxf(mx, S[rr][cidx]);
        #pragma unroll
        for (int m2 = 16; m2 >= 1; m2 >>= 1) mx = fmaxf(mx, __shfl_xor(mx, m2, 32));
        float sum = 0.f;
        for (int cidx = lane; cidx < ncol; cidx += 32) {
            float e2 = __expf(S[rr][cidx] - mx);
            S[rr][cidx] = e2;
            sum += e2;
        }
        #pragma unroll
        for (int m2 = 16; m2 >= 1; m2 >>= 1) sum += __shfl_xor(sum, m2, 32);
        const float inv = 1.0f / sum;
        for (int cidx = lane; cidx < ncol; cidx += 32)
            Pb[rr][cidx] = (__bf16)(S[rr][cidx] * inv);
        if (nkt & 1) {                    // zero-pad to a 32-key boundary
            for (int cidx = ncol + lane; cidx < ncol + 16; cidx += 32)
                Pb[rr][cidx] = (__bf16)0.0f;
        }
    }
    __syncthreads();

    // ---- P @ V : accumulate 16 x 64 output ----
    v8f acc[4];
    #pragma unroll
    for (int j = 0; j < 4; j++) acc[j] = (v8f){0.f,0.f,0.f,0.f,0.f,0.f,0.f,0.f};

    const int nkt2 = (nkt + 1) >> 1;     // 32-key steps
    for (int kk = 0; kk < nkt2; kk++) {
        const int key0 = kk * 32;
        // stage V tile transposed: Vt[dh][key], key = key0 + lane
        {
            const __bf16* vr = vbase + (size_t)(key0 + lane) * DMODEL;
            uint4 raw[8];
            #pragma unroll
            for (int u = 0; u < 8; u++) raw[u] = *(const uint4*)(vr + u * 8);
            const __bf16* hh = (const __bf16*)raw;
            #pragma unroll
            for (int dh = 0; dh < DHEAD; dh++) Vt[dh][lane] = hh[dh];
        }
        __syncthreads();
        Frag16 ap;                        // P tile as A operand 16x32
        ap.q[0] = *(const uint4*)(&Pb[lrow][key0 + lhi * 8]);
        ap.q[1] = *(const uint4*)(&Pb[lrow][key0 + 16 + lhi * 8]);
        #pragma unroll
        for (int j = 0; j < 4; j++) {     // dh groups of 16
            Frag16 bv;                    // B operand: 32 keys x 16 dh
            bv.q[0] = *(const uint4*)(&Vt[j * 16 + lrow][lhi * 8]);
            bv.q[1] = *(const uint4*)(&Vt[j * 16 + lrow][16 + lhi * 8]);
            acc[j] = wmma_bf16(ap.v, bv.v, acc[j]);
        }
        __syncthreads();
    }

    // ---- write y (bf16) ----
    __bf16* ybase = y + (size_t)batch * TLEN * DMODEL + head * DHEAD;
    #pragma unroll
    for (int j = 0; j < 4; j++) {
        #pragma unroll
        for (int e = 0; e < 8; e++) {
            const int m  = e + lhi * 8;
            const int dh = j * 16 + lrow;
            ybase[(size_t)(qt0 + m) * DMODEL + dh] = (__bf16)acc[j][e];
        }
    }
}

// ---------------------------------------------------------------------------
// Host-side launch sequence
// ---------------------------------------------------------------------------
extern "C" void kernel_launch(void* const* d_in, const int* in_sizes, int n_in,
                              void* d_out, int out_size, void* d_ws, size_t ws_size,
                              hipStream_t stream)
{
    (void)in_sizes; (void)n_in; (void)out_size; (void)ws_size;

    const int*   idx    = (const int*)  d_in[0];
    const float* tok    = (const float*)d_in[1];
    const float* pos    = (const float*)d_in[2];
    const float* ln1_w  = (const float*)d_in[3];
    const float* ln1_b  = (const float*)d_in[4];
    const float* wq     = (const float*)d_in[5];
    const float* wk     = (const float*)d_in[6];
    const float* wv     = (const float*)d_in[7];
    const float* wp     = (const float*)d_in[8];
    const float* ln2_w  = (const float*)d_in[9];
    const float* ln2_b  = (const float*)d_in[10];
    const float* fc1_w  = (const float*)d_in[11];
    const float* fc1_b  = (const float*)d_in[12];
    const float* fc2_w  = (const float*)d_in[13];
    const float* fc2_b  = (const float*)d_in[14];
    const float* lnf_w  = (const float*)d_in[15];
    const float* lnf_b  = (const float*)d_in[16];
    const float* head_w = (const float*)d_in[17];

    // workspace layout (44 MiB total)
    char* ws = (char*)d_ws;
    float*  x  = (float*) (ws + 0);                        // 8 MiB  fp32 [2048,1024]
    __bf16* hb = (__bf16*)(ws + (size_t)8  * 1024 * 1024); // 4 MiB  bf16 [2048,1024]
    __bf16* qb = (__bf16*)(ws + (size_t)12 * 1024 * 1024);
    __bf16* kb = (__bf16*)(ws + (size_t)16 * 1024 * 1024);
    __bf16* vb = (__bf16*)(ws + (size_t)20 * 1024 * 1024);
    __bf16* yb = (__bf16*)(ws + (size_t)24 * 1024 * 1024);
    __bf16* gb = (__bf16*)(ws + (size_t)28 * 1024 * 1024); // 16 MiB bf16 [2048,4096]

    // x = tok_emb[idx] + pos_emb
    embed_kernel<<<(MROWS * DMODEL) / 256, 256, 0, stream>>>(idx, tok, pos, x);

    const dim3 gD(MROWS / BM, DMODEL / BN);   // 16 x 4
    const dim3 gF(MROWS / BM, FFN / BN);      // 16 x 16
    const dim3 gH(MROWS / BM, VOCAB / BN);    // 16 x 125
    const dim3 gAtt(TLEN / 16, BATCH * NHEAD);

    for (int l = 0; l < NLAYER; l++) {
        const size_t oDD = (size_t)l * DMODEL * DMODEL;
        const size_t oFD = (size_t)l * FFN * DMODEL;

        layernorm_kernel<<<MROWS, 256, 0, stream>>>(x, ln1_w + l * DMODEL, ln1_b + l * DMODEL, hb, DMODEL);

        gemm_bf16_kernel<<<gD, 256, 0, stream>>>(hb, wq + oDD, nullptr, nullptr,
                                                 nullptr, qb, MROWS, DMODEL, DMODEL, 0);
        gemm_bf16_kernel<<<gD, 256, 0, stream>>>(hb, wk + oDD, nullptr, nullptr,
                                                 nullptr, kb, MROWS, DMODEL, DMODEL, 0);
        gemm_bf16_kernel<<<gD, 256, 0, stream>>>(hb, wv + oDD, nullptr, nullptr,
                                                 nullptr, vb, MROWS, DMODEL, DMODEL, 0);

        attention_kernel<<<gAtt, 32, 0, stream>>>(qb, kb, vb, yb);

        // x = x + y @ Wp^T   (residual fused, in-place safe: 1 block per tile)
        gemm_bf16_kernel<<<gD, 256, 0, stream>>>(yb, wp + oDD, nullptr, x,
                                                 x, nullptr, MROWS, DMODEL, DMODEL, 4);

        layernorm_kernel<<<MROWS, 256, 0, stream>>>(x, ln2_w + l * DMODEL, ln2_b + l * DMODEL, hb, DMODEL);

        // g = gelu(h @ W1^T + b1)
        gemm_bf16_kernel<<<gF, 256, 0, stream>>>(hb, fc1_w + oFD, fc1_b + (size_t)l * FFN, nullptr,
                                                 nullptr, gb, MROWS, FFN, DMODEL, 1 | 2);
        // x = x + g @ W2^T + b2
        gemm_bf16_kernel<<<gD, 256, 0, stream>>>(gb, fc2_w + oFD, fc2_b + (size_t)l * DMODEL, x,
                                                 x, nullptr, MROWS, DMODEL, FFN, 1 | 4);
    }

    // final LN + LM head -> logits fp32 [B,T,V]
    layernorm_kernel<<<MROWS, 256, 0, stream>>>(x, lnf_w, lnf_b, hb, DMODEL);
    gemm_bf16_kernel<<<gH, 256, 0, stream>>>(hb, head_w, nullptr, nullptr,
                                             (float*)d_out, nullptr, MROWS, VOCAB, DMODEL, 0);
}
